// HybridDGNN_10393820856801
// MI455X (gfx1250) — matrix operand order repaired and checked by reading, not compiled, not run on hardware
//
#include <hip/hip_runtime.h>
#include <hip/hip_bf16.h>

typedef __attribute__((ext_vector_type(16))) _Float16 v16h;
typedef __attribute__((ext_vector_type(8)))  float    v8f;
typedef __attribute__((ext_vector_type(4)))  unsigned uint4v;
typedef __attribute__((ext_vector_type(8)))  int      int8v;
typedef __attribute__((ext_vector_type(4)))  int      int4v;
typedef _Float16 f16;

#define N_PTS 8192
#define KNN   20
#define FEAT  64
#define NCLS  13

// -------------------------------------------------------------------------
// Tensor Data Mover: 1D contiguous tile (global f32 -> LDS). D# packing per
// CDNA5 ISA ch.8 (group0: count/lds/global/type, group1: dims+strides).
// 6-arg builtin form (clang-23 / therock-10.0): (g0, g1, g2, g3, g4, cpol).
// Issued wave-uniform; completion via s_wait_tensorcnt.
// -------------------------------------------------------------------------
__device__ __forceinline__ void tdm_load_1d(const void* gsrc, unsigned lds_off,
                                            unsigned tile_elems, unsigned tensor_elems) {
    const unsigned long long ga = (unsigned long long)gsrc;
    uint4v g0;
    g0.x = 1u;                                             // count=1, user descriptor
    g0.y = lds_off;                                        // LDS byte address
    g0.z = (unsigned)(ga & 0xFFFFFFFFull);                 // global_addr[31:0]
    g0.w = (unsigned)((ga >> 32) & 0x01FFFFFFull)          // global_addr[56:32]
         | (2u << 30);                                     // type=2 ("image")
    int8v g1;
    g1[0] = (int)(2u << 16);                               // data_size=4B, no multicast
    g1[1] = (int)((tensor_elems & 0xFFFFu) << 16);         // tensor_dim0[15:0]
    g1[2] = (int)(((tensor_elems >> 16) & 0xFFFFu)         // tensor_dim0[31:16]
         | (1u << 16));                                    // tensor_dim1[15:0] = 1
    g1[3] = (int)((tile_elems & 0xFFFFu) << 16);           // tile_dim0
    g1[4] = 1;                                             // tile_dim1=1, tile_dim2=0
    g1[5] = (int)tensor_elems;                             // tensor_dim0_stride[31:0]
    g1[6] = 0;
    g1[7] = 0;
    int4v z4 = {0, 0, 0, 0};                               // <=2D: groups 2/3 unused
    int8v z8 = {0, 0, 0, 0, 0, 0, 0, 0};
    __builtin_amdgcn_tensor_load_to_lds(g0, g1, z4, z4, z8, 0);
}

// -------------------------------------------------------------------------
// kNN: one thread per query row; column features tiled through LDS by the
// TDM. Unsorted top-K kept in registers via fully unrolled replace-worst.
// -------------------------------------------------------------------------
template<int C>
__global__ void knn_kernel(const float* __restrict__ x, int* __restrict__ idx_out) {
    const int i = blockIdx.x * blockDim.x + threadIdx.x;   // query row
    constexpr int TJ = 128;
    __shared__ float tile[TJ * C];

    float xi[C];
    #pragma unroll
    for (int c = 0; c < C; ++c) xi[c] = x[i * C + c];

    float bd[KNN]; int bi[KNN];
    #pragma unroll
    for (int s = 0; s < KNN; ++s) { bd[s] = 3.4e38f; bi[s] = 0; }
    float worst = 3.4e38f; int worstSlot = 0;

    const unsigned lds_off = (unsigned)(unsigned long long)(tile);

    for (int jb = 0; jb < N_PTS; jb += TJ) {
        __syncthreads();                                   // tile no longer read
        if (threadIdx.x < 32) {                            // wave 0 drives the TDM
            tdm_load_1d(x + (size_t)jb * C, lds_off, TJ * C, N_PTS * C);
            __builtin_amdgcn_s_wait_tensorcnt(0);
        }
        __syncthreads();                                   // tile ready for all waves
        for (int jj = 0; jj < TJ; ++jj) {
            const int j = jb + jj;
            float d = 0.f;
            #pragma unroll
            for (int c = 0; c < C; ++c) {
                float t = xi[c] - tile[jj * C + c];
                d += t * t;
            }
            if (j == i) continue;                          // exclude self
            if (d < worst) {
                #pragma unroll
                for (int s = 0; s < KNN; ++s)
                    if (s == worstSlot) { bd[s] = d; bi[s] = j; }
                worst = -1.f;
                #pragma unroll
                for (int s = 0; s < KNN; ++s)
                    if (bd[s] > worst) { worst = bd[s]; worstSlot = s; }
            }
        }
    }
    #pragma unroll
    for (int s = 0; s < KNN; ++s) idx_out[i * KNN + s] = bi[s];
}

// -------------------------------------------------------------------------
// EdgeConv: block = one point, 64 threads = 64 output features.
// h_ij = relu(b + xi.Wtop + (xj-xi).Wbot) ; out = max_j h_ij.
// xi.Wtop is j-invariant -> hoisted. Gather staged in LDS once per point.
// -------------------------------------------------------------------------
template<int C>
__global__ void edgeconv_kernel(const float* __restrict__ x, const int* __restrict__ knn,
                                const float* __restrict__ w, const float* __restrict__ bias,
                                float* __restrict__ out) {
    const int i = blockIdx.x;          // point index
    const int o = threadIdx.x;         // output feature (blockDim == FEAT)
    __shared__ float sxi[C];
    __shared__ float sxj[KNN * C];

    for (int t = threadIdx.x; t < C; t += blockDim.x) sxi[t] = x[i * C + t];
    for (int t = threadIdx.x; t < KNN * C; t += blockDim.x) {
        const int j = knn[i * KNN + t / C];
        sxj[t] = x[j * C + (t % C)];
    }
    __syncthreads();

    float base = bias[o];
    #pragma unroll
    for (int c = 0; c < C; ++c) base += sxi[c] * w[c * FEAT + o];

    float best = 0.f;                  // relu output >= 0 => 0 is a safe init
    for (int j = 0; j < KNN; ++j) {
        float acc = base;
        #pragma unroll
        for (int c = 0; c < C; ++c)
            acc += (sxj[j * C + c] - sxi[c]) * w[(C + c) * FEAT + o];
        acc = acc > 0.f ? acc : 0.f;
        best = acc > best ? acc : best;
    }
    out[i * FEAT + o] = best;
}

// -------------------------------------------------------------------------
// f16 staging: weight cast+transpose (Bt[N][K]) and activation concat.
// -------------------------------------------------------------------------
__global__ void cast_transpose_f16_kernel(const float* __restrict__ B, f16* __restrict__ Bt,
                                          int K, int N) {
    const int t = blockIdx.x * blockDim.x + threadIdx.x;
    if (t >= K * N) return;
    const int k = t / N, n = t % N;
    Bt[(size_t)n * K + k] = (f16)B[t];
}

__global__ void concat_cast_kernel(const float* __restrict__ x1, const float* __restrict__ x2,
                                   const float* __restrict__ x3, f16* __restrict__ cat) {
    const int t = blockIdx.x * blockDim.x + threadIdx.x;   // N*192 elements
    const int i = t / (3 * FEAT), c = t % (3 * FEAT);
    float v;
    if      (c < FEAT)     v = x1[i * FEAT + c];
    else if (c < 2 * FEAT) v = x2[i * FEAT + c - FEAT];
    else                   v = x3[i * FEAT + c - 2 * FEAT];
    cat[t] = (f16)v;
}

// -------------------------------------------------------------------------
// WMMA GEMM: D[M,N] = act(A[M,K] * B[K,N] + bias) with B pre-transposed to
// Bt[N][K]. One wave computes a 16x64 strip (4 accumulators) so each A
// fragment feeds 4 v_wmma_f32_16x16x32_f16. All fragment loads are b128.
// Fragment packing: contiguous k = 16*hi + s in both A and B (wave32).
// -------------------------------------------------------------------------
union Frag { v16h v; uint4v q[2]; f16 h[16]; };

template<typename OutT, bool RELU>
__global__ void gemm_wmma_kernel(const f16* __restrict__ A, const f16* __restrict__ Bt,
                                 const float* __restrict__ bias, OutT* __restrict__ D,
                                 int M, int N, int Kd) {
    const int lane    = threadIdx.x & 31;
    const int wave    = threadIdx.x >> 5;
    const int groupsN = N >> 6;                            // 64 columns per wave
    const int g       = blockIdx.x * (blockDim.x >> 5) + wave;
    const int tm      = (g / groupsN) << 4;
    const int tn      = (g % groupsN) << 6;
    if (tm >= M) return;               // wave-uniform: EXEC stays all-ones for WMMA

    const int hi  = lane >> 4;         // half-wave select
    const int l16 = lane & 15;

    v8f acc0 = {}, acc1 = {}, acc2 = {}, acc3 = {};
    const f16* arow  = A  + (size_t)(tm + l16) * Kd + hi * 16;
    const f16* brow0 = Bt + (size_t)(tn + l16) * Kd + hi * 16;
    const f16* brow1 = brow0 + (size_t)16 * Kd;
    const f16* brow2 = brow0 + (size_t)32 * Kd;
    const f16* brow3 = brow0 + (size_t)48 * Kd;

    for (int k0 = 0; k0 < Kd; k0 += 32) {
        Frag a, b0, b1, b2, b3;
        __builtin_prefetch(arow + k0 + 64, 0, 0);          // global_prefetch_b8
        a.q[0]  = *(const uint4v*)(arow  + k0);            // K 0..7   (+hi*16)
        a.q[1]  = *(const uint4v*)(arow  + k0 + 8);        // K 8..15  (+hi*16)
        b0.q[0] = *(const uint4v*)(brow0 + k0);            // K 0..7   (+hi*16)
        b0.q[1] = *(const uint4v*)(brow0 + k0 + 8);        // K 8..15  (+hi*16)
        b1.q[0] = *(const uint4v*)(brow1 + k0);
        b1.q[1] = *(const uint4v*)(brow1 + k0 + 8);
        b2.q[0] = *(const uint4v*)(brow2 + k0);
        b2.q[1] = *(const uint4v*)(brow2 + k0 + 8);
        b3.q[0] = *(const uint4v*)(brow3 + k0);
        b3.q[1] = *(const uint4v*)(brow3 + k0 + 8);
        acc0 = __builtin_amdgcn_wmma_f32_16x16x32_f16(false, a.v, false, b0.v, (short)0, acc0, false, false);
        acc1 = __builtin_amdgcn_wmma_f32_16x16x32_f16(false, a.v, false, b1.v, (short)0, acc1, false, false);
        acc2 = __builtin_amdgcn_wmma_f32_16x16x32_f16(false, a.v, false, b2.v, (short)0, acc2, false, false);
        acc3 = __builtin_amdgcn_wmma_f32_16x16x32_f16(false, a.v, false, b3.v, (short)0, acc3, false, false);
    }

    #pragma unroll
    for (int r = 0; r < 8; ++r) {
        const int m = tm + r + hi * 8;
        OutT* drow = D + (size_t)m * N;
        float v;
        int n;
        n = tn + l16;      v = acc0[r] + bias[n]; if (RELU) v = v > 0.f ? v : 0.f; drow[n] = (OutT)v;
        n = tn + 16 + l16; v = acc1[r] + bias[n]; if (RELU) v = v > 0.f ? v : 0.f; drow[n] = (OutT)v;
        n = tn + 32 + l16; v = acc2[r] + bias[n]; if (RELU) v = v > 0.f ? v : 0.f; drow[n] = (OutT)v;
        n = tn + 48 + l16; v = acc3[r] + bias[n]; if (RELU) v = v > 0.f ? v : 0.f; drow[n] = (OutT)v;
    }
}

// -------------------------------------------------------------------------
// Final 128 -> 13 layer fused with log_softmax (tiny K, VALU).
// -------------------------------------------------------------------------
__global__ void final_logsoftmax_kernel(const float* __restrict__ h, const float* __restrict__ w,
                                        const float* __restrict__ bias, float* __restrict__ out) {
    const int i = blockIdx.x * blockDim.x + threadIdx.x;
    if (i >= N_PTS) return;
    float lg[NCLS];
    float mx = -3.4e38f;
    for (int o = 0; o < NCLS; ++o) {
        float a = bias[o];
        for (int c = 0; c < 128; ++c) a += h[i * 128 + c] * w[c * NCLS + o];
        lg[o] = a;
        mx = a > mx ? a : mx;
    }
    float s = 0.f;
    #pragma unroll
    for (int o = 0; o < NCLS; ++o) s += __expf(lg[o] - mx);
    const float ls = __logf(s);
    #pragma unroll
    for (int o = 0; o < NCLS; ++o) out[i * NCLS + o] = lg[o] - mx - ls;
}

// -------------------------------------------------------------------------
// Orchestration
// -------------------------------------------------------------------------
extern "C" void kernel_launch(void* const* d_in, const int* in_sizes, int n_in,
                              void* d_out, int out_size, void* d_ws, size_t ws_size,
                              hipStream_t stream) {
    const float* x   = (const float*)d_in[0];
    const float* w1  = (const float*)d_in[1];
    const float* b1  = (const float*)d_in[2];
    const float* w2  = (const float*)d_in[3];
    const float* b2  = (const float*)d_in[4];
    const float* w3  = (const float*)d_in[5];
    const float* b3  = (const float*)d_in[6];
    const float* wl1 = (const float*)d_in[7];
    const float* bl1 = (const float*)d_in[8];
    const float* wm1 = (const float*)d_in[9];
    const float* bm1 = (const float*)d_in[10];
    const float* wm2 = (const float*)d_in[11];
    const float* bm2 = (const float*)d_in[12];
    const float* wm3 = (const float*)d_in[13];
    const float* bm3 = (const float*)d_in[14];
    float* out = (float*)d_out;

    // bump allocator over d_ws
    char* wsp = (char*)d_ws;
    auto alloc = [&](size_t bytes) -> void* {
        void* p = (void*)wsp;
        wsp += (bytes + 255) & ~(size_t)255;
        return p;
    };
    int*   idx    = (int*)  alloc((size_t)N_PTS * KNN * 4);
    float* x1     = (float*)alloc((size_t)N_PTS * FEAT * 4);
    float* x2     = (float*)alloc((size_t)N_PTS * FEAT * 4);
    float* x3     = (float*)alloc((size_t)N_PTS * FEAT * 4);
    f16*   cat16  = (f16*)  alloc((size_t)N_PTS * 3 * FEAT * 2);
    f16*   wl1_t  = (f16*)  alloc((size_t)192 * 1024 * 2);   // [1024][192]
    f16*   wm1_t  = (f16*)  alloc((size_t)1024 * 256 * 2);   // [256][1024]
    f16*   wm2_t  = (f16*)  alloc((size_t)256 * 128 * 2);    // [128][256]
    f16*   h1_16  = (f16*)  alloc((size_t)N_PTS * 1024 * 2);
    f16*   h2_16  = (f16*)  alloc((size_t)N_PTS * 256 * 2);
    float* h3     = (float*)alloc((size_t)N_PTS * 128 * 4);

    // ---- EdgeConv stack (dynamic graph: kNN recomputed on each feature map)
    knn_kernel<3>   <<<N_PTS / 256, 256, 0, stream>>>(x, idx);
    edgeconv_kernel<3><<<N_PTS, FEAT, 0, stream>>>(x, idx, w1, b1, x1);

    knn_kernel<FEAT><<<N_PTS / 256, 256, 0, stream>>>(x1, idx);
    edgeconv_kernel<FEAT><<<N_PTS, FEAT, 0, stream>>>(x1, idx, w2, b2, x2);

    knn_kernel<FEAT><<<N_PTS / 256, 256, 0, stream>>>(x2, idx);
    edgeconv_kernel<FEAT><<<N_PTS, FEAT, 0, stream>>>(x2, idx, w3, b3, x3);

    // ---- Stage f16 operands (weights cast + transposed for b128 frag loads)
    {
        int n = N_PTS * 3 * FEAT;
        concat_cast_kernel<<<(n + 255) / 256, 256, 0, stream>>>(x1, x2, x3, cat16);
    }
    { int n = 192 * 1024; cast_transpose_f16_kernel<<<(n + 255) / 256, 256, 0, stream>>>(wl1, wl1_t, 192, 1024); }
    { int n = 1024 * 256; cast_transpose_f16_kernel<<<(n + 255) / 256, 256, 0, stream>>>(wm1, wm1_t, 1024, 256); }
    { int n = 256 * 128;  cast_transpose_f16_kernel<<<(n + 255) / 256, 256, 0, stream>>>(wm2, wm2_t, 256, 128); }

    // ---- Dense stack on the matrix pipes (wave32 WMMA, 16x64 strip / wave)
    // h1 = relu(cat @ wl1 + bl1)   : M=8192 N=1024 K=192
    {
        int waves = (N_PTS / 16) * (1024 / 64);
        gemm_wmma_kernel<f16, true><<<waves / 8, 256, 0, stream>>>(cat16, wl1_t, bl1, h1_16, N_PTS, 1024, 192);
    }
    // h2 = relu(h1 @ wm1 + bm1)    : M=8192 N=256 K=1024
    {
        int waves = (N_PTS / 16) * (256 / 64);
        gemm_wmma_kernel<f16, true><<<waves / 8, 256, 0, stream>>>(h1_16, wm1_t, bm1, h2_16, N_PTS, 256, 1024);
    }
    // h3 = relu(h2 @ wm2 + bm2)    : M=8192 N=128 K=256
    {
        int waves = (N_PTS / 16) * (128 / 64);
        gemm_wmma_kernel<float, true><<<waves / 8, 256, 0, stream>>>(h2_16, wm2_t, bm2, h3, N_PTS, 128, 256);
    }

    // ---- logits + log_softmax
    final_logsoftmax_kernel<<<N_PTS / 256, 256, 0, stream>>>(h3, wm3, bm3, out);
}
